// GNN_36094905155666
// MI455X (gfx1250) — matrix operand order repaired and checked by reading, not compiled, not run on hardware
//
#include <hip/hip_runtime.h>
#include <hip/hip_bf16.h>
#include <math.h>

// ---------------------------------------------------------------------------
// Problem constants (from reference)
// ---------------------------------------------------------------------------
#define BGRAPH 128
#define N_PER  256
#define F_IN   128
#define HID    128
#define HEADS  4
#define HC     (HEADS * HID)    // 512
#define N0     (BGRAPH * N_PER) // 32768
#define DEG    8
#define ETOT   (N0 * DEG)       // 262144
#define EPG    (N_PER * DEG)    // 2048 edges per graph (fixed slab)
#define FC1    256
#define FC2    128
#define NEG_SLOPE 0.2f

typedef __attribute__((ext_vector_type(16))) __bf16 v16bf;
typedef __attribute__((ext_vector_type(4)))  __bf16 v4bf;
typedef __attribute__((ext_vector_type(8)))  float  v8f;
typedef __attribute__((ext_vector_type(4)))  float  v4f;

// ---------------------------------------------------------------------------
// WMMA GEMM: C[M,N] = A[M,K] @ W[N,K]^T (+bias) (+relu)
// fp32 inputs converted to bf16 in LDS (float4 loads -> v_cvt_pk_bf16_f32 ->
// ds_store_b64), fp32 accumulate via v_wmma_f32_16x16x32_bf16.
// Requires M%128==0, N%64==0, K%32==0 (true for every call site).
// Block = 128 threads (4 waves). Each wave: 32 rows x 64 cols
//   = 2 A-fragments x 4 B-fragments = 8 WMMAs per K-step.
// Staging loops have compile-time trip counts (no exec-mask guards).
// ---------------------------------------------------------------------------
#define TM 128
#define TN 64
#define TK 32

__global__ __launch_bounds__(128)
void gemm_wmma(const float* __restrict__ A, const float* __restrict__ W,
               const float* __restrict__ bias, float* __restrict__ C,
               int M, int N, int K, int relu)
{
    __shared__ __bf16 sA[TM][TK];   // 8 KB
    __shared__ __bf16 sB[TN][TK];   // 4 KB

    const int tid  = threadIdx.x;
    const int wave = tid >> 5;
    const int lane = tid & 31;
    const int half = lane >> 4;     // which 16-lane half
    const int l16  = lane & 15;
    const int rowBase = blockIdx.y * TM;
    const int colBase = blockIdx.x * TN;

    // Per-thread staging coordinates (uniform trip counts, no guards):
    // each thread owns one float4 chunk per iteration; 8 iters for A, 4 for B.
    const int chunkRowStep = 128 / (TK / 4);    // 16 rows per 128-thread sweep
    const int myRow = tid >> 3;                 // 0..15
    const int myC4  = (tid & 7) * 4;            // 0,4,...,28

    v8f acc[2][4] = {};

    for (int k0 = 0; k0 < K; k0 += TK) {
        // Stage A tile: 128 rows x 32 cols, 8 sweeps of 16 rows.
        #pragma unroll
        for (int it = 0; it < TM / chunkRowStep; ++it) {
            int r = myRow + it * chunkRowStep;
            v4f a = *(const v4f*)&A[(size_t)(rowBase + r) * K + k0 + myC4];
            v4bf b;
            b[0] = (__bf16)a[0]; b[1] = (__bf16)a[1];
            b[2] = (__bf16)a[2]; b[3] = (__bf16)a[3];
            *(v4bf*)&sA[r][myC4] = b;           // ds_store_b64
        }
        // Stage B tile: 64 rows x 32 cols, 4 sweeps of 16 rows.
        #pragma unroll
        for (int it = 0; it < TN / chunkRowStep; ++it) {
            int r = myRow + it * chunkRowStep;
            v4f a = *(const v4f*)&W[(size_t)(colBase + r) * K + k0 + myC4];
            v4bf b;
            b[0] = (__bf16)a[0]; b[1] = (__bf16)a[1];
            b[2] = (__bf16)a[2]; b[3] = (__bf16)a[3];
            *(v4bf*)&sB[r][myC4] = b;
        }
        __syncthreads();

        // B fragments (32x16, ISA 7.12.2): lane = column; VGPR j holds K pairs
        // {2j,2j+1} (+8 for lanes 16-31); VGPRs 4-7 are K+16. Hoisted and
        // reused by both A row-tiles.
        v16bf bfr[4];
        #pragma unroll
        for (int t = 0; t < 4; t++) {
            int bn = t * 16 + l16;
            #pragma unroll
            for (int j = 0; j < 8; j++) {
                int kk = (j < 4) ? (2 * j + half * 8) : (16 + 2 * (j - 4) + half * 8);
                bfr[t][2 * j]     = sB[bn][kk];
                bfr[t][2 * j + 1] = sB[bn][kk + 1];
            }
        }

        #pragma unroll
        for (int mi = 0; mi < 2; mi++) {
            // A fragment (16x32): lanes 0-15 row m, VGPR i holds K pairs
            // {2i,2i+1} (+8 for lanes 16-31); VGPRs 4-7 are K+16.
            v16bf afrag;
            int am = wave * 32 + mi * 16 + l16;
            #pragma unroll
            for (int i = 0; i < 8; i++) {
                int kb = (i < 4) ? (2 * i + half * 8)
                                 : (16 + 2 * (i - 4) + half * 8);
                afrag[2 * i]     = sA[am][kb];
                afrag[2 * i + 1] = sA[am][kb + 1];
            }
            #pragma unroll
            for (int t = 0; t < 4; t++) {
                acc[mi][t] = __builtin_amdgcn_wmma_f32_16x16x32_bf16(
                    false, afrag, false, bfr[t], (short)0, acc[mi][t],
                    false, false);
            }
        }
        __syncthreads();
    }

    // C/D layout: element r of v8f at lane L -> row r + 8*(L/16), col L%16.
    #pragma unroll
    for (int mi = 0; mi < 2; mi++) {
        #pragma unroll
        for (int t = 0; t < 4; t++) {
            #pragma unroll
            for (int r = 0; r < 8; r++) {
                int m = rowBase + wave * 32 + mi * 16 + r + 8 * half;
                int n = colBase + t * 16 + l16;
                float v = acc[mi][t][r];
                if (bias) v += bias[n];
                if (relu) v = fmaxf(v, 0.0f);
                C[(size_t)m * N + n] = v;
            }
        }
    }
}

// ---------------------------------------------------------------------------
// alpha_src/alpha_dst per node: as[n,h] = sum_c h[n,h*128+c]*a_src[h,c]
// One block (4 waves) per node, wave = head, shfl reduce.
// ---------------------------------------------------------------------------
__global__ __launch_bounds__(128)
void alpha_kernel(const float* __restrict__ h, const float* __restrict__ a_src,
                  const float* __restrict__ a_dst,
                  float* __restrict__ as_n, float* __restrict__ ad_n)
{
    int node = blockIdx.x;
    int head = threadIdx.x >> 5;
    int lane = threadIdx.x & 31;
    const float* hp = h + (size_t)node * HC + head * HID;
    float s = 0.f, d = 0.f;
    for (int c = lane; c < HID; c += 32) {
        float v = hp[c];
        s += v * a_src[head * HID + c];
        d += v * a_dst[head * HID + c];
    }
    for (int off = 16; off; off >>= 1) {
        s += __shfl_xor(s, off, 32);
        d += __shfl_xor(d, off, 32);
    }
    if (lane == 0) { as_n[node * HEADS + head] = s; ad_n[node * HEADS + head] = d; }
}

// ---------------------------------------------------------------------------
// CSR-by-destination build (per-graph fixed slab of EPG entries).
// ---------------------------------------------------------------------------
__global__ void fill_int(int* __restrict__ p, int v, int n)
{
    int i = blockIdx.x * blockDim.x + threadIdx.x;
    if (i < n) p[i] = v;
}

__global__ void csr_count(const int* __restrict__ dst, const int* __restrict__ em,
                          int E, int* __restrict__ cnt)
{
    int e = blockIdx.x * blockDim.x + threadIdx.x;
    if (e < E && em[e]) atomicAdd(&cnt[dst[e]], 1);
}

__global__ __launch_bounds__(256)
void csr_scan(const int* __restrict__ cnt, int* __restrict__ offs,
              int* __restrict__ cursor, int n)
{
    __shared__ int s[256];
    int g = blockIdx.x, t = threadIdx.x;
    int v = (t < n) ? cnt[g * n + t] : 0;
    s[t] = v;
    __syncthreads();
    for (int off = 1; off < 256; off <<= 1) {      // Hillis-Steele inclusive
        int x = (t >= off) ? s[t - off] : 0;
        __syncthreads();
        s[t] += x;
        __syncthreads();
    }
    if (t < n) {
        int o = g * EPG + (s[t] - v);              // exclusive, graph slab base
        offs[g * n + t] = o;
        cursor[g * n + t] = o;
    }
}

__global__ void csr_fill(const int* __restrict__ dst, const int* __restrict__ em,
                         int E, int* __restrict__ cursor, int* __restrict__ elist)
{
    int e = blockIdx.x * blockDim.x + threadIdx.x;
    if (e < E && em[e]) {
        int p = atomicAdd(&cursor[dst[e]], 1);
        elist[p] = e;
    }
}

// ---------------------------------------------------------------------------
// GAT softmax + aggregation, gather form (no float atomics).
// One 128-thread block per destination node.
// out[n,ch] = (sum_e w_e * h[src_e,ch] + w_self*h[n,ch]) / denom + att_b[ch]
// ---------------------------------------------------------------------------
#define WCAP 256
__global__ __launch_bounds__(128)
void gat_aggregate(const float* __restrict__ h, const float* __restrict__ as_n,
                   const float* __restrict__ ad_n, const float* __restrict__ att_b,
                   const int* __restrict__ src, const int* __restrict__ offs,
                   const int* __restrict__ cnt, const int* __restrict__ elist,
                   float* __restrict__ out)
{
    __shared__ float sm[HEADS], sden[HEADS], swself[HEADS];
    __shared__ float wbuf[HEADS][WCAP];

    int node = blockIdx.x;
    int tid  = threadIdx.x;
    int head = tid >> 5;
    int lane = tid & 31;
    int deg   = cnt[node];
    int start = offs[node];
    float adn = ad_n[node * HEADS + head];

    // phase 1 (per head-wave): max logit, then exp-sum
    float mx = -__builtin_inff();
    for (int i = lane; i < deg; i += 32) {
        int e = elist[start + i];
        float v = as_n[src[e] * HEADS + head] + adn;
        v = (v >= 0.f) ? v : NEG_SLOPE * v;
        mx = fmaxf(mx, v);
    }
    for (int off = 16; off; off >>= 1) mx = fmaxf(mx, __shfl_xor(mx, off, 32));
    float eself = as_n[node * HEADS + head] + adn;
    eself = (eself >= 0.f) ? eself : NEG_SLOPE * eself;
    mx = fmaxf(mx, eself);

    float sum = 0.f;
    for (int i = lane; i < deg; i += 32) {
        int e = elist[start + i];
        float v = as_n[src[e] * HEADS + head] + adn;
        v = (v >= 0.f) ? v : NEG_SLOPE * v;
        float w = __expf(v - mx);
        if (i < WCAP) wbuf[head][i] = w;
        sum += w;
    }
    for (int off = 16; off; off >>= 1) sum += __shfl_xor(sum, off, 32);
    float ws = __expf(eself - mx);
    if (lane == 0) { sm[head] = mx; sden[head] = sum + ws; swself[head] = ws; }
    __syncthreads();

    // phase 2: thread tid handles channel (j, tid) for each head j
    for (int j = 0; j < HEADS; j++) {
        int ch = j * HID + tid;
        float acc = swself[j] * h[(size_t)node * HC + ch];
        float mj = sm[j], adj = ad_n[node * HEADS + j];
        for (int i = 0; i < deg; i++) {
            int e = elist[start + i];
            int s = src[e];
            float w;
            if (i < WCAP) {
                w = wbuf[j][i];
            } else {
                float v = as_n[s * HEADS + j] + adj;
                v = (v >= 0.f) ? v : NEG_SLOPE * v;
                w = __expf(v - mj);
            }
            acc += w * h[(size_t)s * HC + ch];
        }
        out[(size_t)node * HC + ch] = acc / sden[j] + att_b[ch];
    }
}

// ---------------------------------------------------------------------------
// TopK pooling pieces
// ---------------------------------------------------------------------------
__global__ __launch_bounds__(128)
void pnorm_kernel(const float* __restrict__ p, float* __restrict__ out)
{
    __shared__ float s[128];
    int t = threadIdx.x;
    float v = p[t];
    s[t] = v * v;
    __syncthreads();
    for (int off = 64; off; off >>= 1) {
        if (t < off) s[t] += s[t + off];
        __syncthreads();
    }
    if (t == 0) out[0] = sqrtf(s[0]);
}

__global__ __launch_bounds__(128)
void score_kernel(const float* __restrict__ x, const float* __restrict__ p,
                  const float* __restrict__ pnorm, float* __restrict__ score, int N)
{
    int node = blockIdx.x * 4 + (threadIdx.x >> 5);
    int lane = threadIdx.x & 31;
    if (node >= N) return;
    float s = 0.f;
    for (int c = lane; c < HID; c += 32) s += x[(size_t)node * HID + c] * p[c];
    for (int off = 16; off; off >>= 1) s += __shfl_xor(s, off, 32);
    if (lane == 0) score[node] = s / pnorm[0];
}

// Per-graph bitonic sort descending (score, then lower index) -> perm, new_pos
__global__ __launch_bounds__(256)
void topk_kernel(const float* __restrict__ score, int n, int k,
                 int* __restrict__ perm, int* __restrict__ new_pos)
{
    __shared__ float sv[256];
    __shared__ int   si[256];
    int g = blockIdx.x, t = threadIdx.x;
    sv[t] = (t < n) ? score[g * n + t] : -__builtin_inff();
    si[t] = t;
    if (t < n) new_pos[g * n + t] = -1;
    __syncthreads();
    for (int size = 2; size <= 256; size <<= 1) {
        for (int stride = size >> 1; stride; stride >>= 1) {
            int p = t ^ stride;
            if (p > t) {
                float a = sv[t], b = sv[p];
                int ia = si[t], ib = si[p];
                bool dirDesc  = ((t & size) == 0);
                bool aFirst   = (a > b) || (a == b && ia < ib);
                if (dirDesc ? !aFirst : aFirst) {
                    sv[t] = b; sv[p] = a; si[t] = ib; si[p] = ia;
                }
            }
            __syncthreads();
        }
    }
    if (t < k) {
        int local = si[t];
        perm[g * k + t] = g * n + local;
        new_pos[g * n + local] = g * k + t;
    }
}

__global__ void gather_scale(const float* __restrict__ x, const float* __restrict__ score,
                             const int* __restrict__ perm, float* __restrict__ xnew, int NK)
{
    int i = blockIdx.x * blockDim.x + threadIdx.x;
    if (i >= NK * HID) return;
    int row = i >> 7, c = i & (HID - 1);
    int s = perm[row];
    xnew[i] = x[(size_t)s * HID + c] * tanhf(score[s]);
}

__global__ void edge_remap(const int* __restrict__ src, const int* __restrict__ dst,
                           const int* __restrict__ em, const int* __restrict__ new_pos,
                           int E, int* __restrict__ src2, int* __restrict__ dst2,
                           int* __restrict__ em2)
{
    int e = blockIdx.x * blockDim.x + threadIdx.x;
    if (e >= E) return;
    int m = em[e];
    int s = -1, d = -1;
    if (m) { s = new_pos[src[e]]; d = new_pos[dst[e]]; }
    int keep = m && s >= 0 && d >= 0;
    src2[e] = keep ? s : 0;
    dst2[e] = keep ? d : 0;
    em2[e]  = keep;
}

// mean/max readout per graph over k pooled nodes -> accumulate into z[g,256]
__global__ __launch_bounds__(128)
void readout_kernel(const float* __restrict__ xnew, int k, float* __restrict__ z, int init)
{
    int g = blockIdx.x, c = threadIdx.x;
    float sum = 0.f, mx = -__builtin_inff();
    for (int r = 0; r < k; r++) {
        float v = xnew[(size_t)(g * k + r) * HID + c];
        sum += v;
        mx = fmaxf(mx, v);
    }
    float mean = sum / (float)k;
    if (init) { z[g * 2 * HID + c] = mean; z[g * 2 * HID + HID + c] = mx; }
    else      { z[g * 2 * HID + c] += mean; z[g * 2 * HID + HID + c] += mx; }
}

// final FC3 (N=1): wave-dot per graph
__global__ __launch_bounds__(32)
void fc3_kernel(const float* __restrict__ z2, const float* __restrict__ W3,
                const float* __restrict__ b3, float* __restrict__ out)
{
    int g = blockIdx.x, lane = threadIdx.x;
    float s = 0.f;
    for (int c = lane; c < FC2; c += 32) s += z2[g * FC2 + c] * W3[c];
    for (int off = 16; off; off >>= 1) s += __shfl_xor(s, off, 32);
    if (lane == 0) out[g] = s + b3[0];
}

// ---------------------------------------------------------------------------
// Orchestration
// ---------------------------------------------------------------------------
extern "C" void kernel_launch(void* const* d_in, const int* in_sizes, int n_in,
                              void* d_out, int out_size, void* d_ws, size_t ws_size,
                              hipStream_t stream)
{
    (void)in_sizes; (void)n_in; (void)out_size; (void)ws_size;
    // JAX pytree flattening order (dict keys sorted):
    // 0 batch_index, 1 edge, 2 edge_index,
    // per block (sorted): att_W, att_a_dst, att_a_src, att_b, lin_W, lin_b, pool_p
    // 3..9 block1, 10..16 block2, 17..23 block3,
    // 24 fc_W1, 25 fc_W2, 26 fc_W3, 27 fc_b1, 28 fc_b2, 29 fc_b3, 30 x
    const int* edge_index = (const int*)d_in[2];
    const float* fc_W1 = (const float*)d_in[24];
    const float* fc_W2 = (const float*)d_in[25];
    const float* fc_W3 = (const float*)d_in[26];
    const float* fc_b1 = (const float*)d_in[27];
    const float* fc_b2 = (const float*)d_in[28];
    const float* fc_b3 = (const float*)d_in[29];
    const float* x_in  = (const float*)d_in[30];

    // workspace carve
    char* w = (char*)d_ws;
    auto carve = [&](size_t bytes) -> void* {
        void* p = (void*)w;
        w += (bytes + 255) & ~(size_t)255;
        return p;
    };
    float* h      = (float*)carve(sizeof(float) * (size_t)N0 * HC);
    float* gatout = (float*)carve(sizeof(float) * (size_t)N0 * HC);
    float* linout = (float*)carve(sizeof(float) * (size_t)N0 * HID);
    float* xA     = (float*)carve(sizeof(float) * (size_t)N0 * HID);
    float* xB     = (float*)carve(sizeof(float) * (size_t)N0 * HID);
    float* as_n   = (float*)carve(sizeof(float) * (size_t)N0 * HEADS);
    float* ad_n   = (float*)carve(sizeof(float) * (size_t)N0 * HEADS);
    float* score  = (float*)carve(sizeof(float) * N0);
    float* pnorm  = (float*)carve(256);
    float* z      = (float*)carve(sizeof(float) * BGRAPH * 2 * HID);
    float* fc1o   = (float*)carve(sizeof(float) * BGRAPH * FC1);
    float* fc2o   = (float*)carve(sizeof(float) * BGRAPH * FC2);
    int* perm     = (int*)carve(sizeof(int) * BGRAPH * N_PER);
    int* new_pos  = (int*)carve(sizeof(int) * N0);
    int* cnt      = (int*)carve(sizeof(int) * N0);
    int* offs     = (int*)carve(sizeof(int) * N0);
    int* cursor   = (int*)carve(sizeof(int) * N0);
    int* elist    = (int*)carve(sizeof(int) * ETOT);
    int* srcBuf[2], *dstBuf[2], *emBuf[2];
    srcBuf[0] = (int*)carve(sizeof(int) * ETOT);
    dstBuf[0] = (int*)carve(sizeof(int) * ETOT);
    emBuf[0]  = (int*)carve(sizeof(int) * ETOT);
    srcBuf[1] = (int*)carve(sizeof(int) * ETOT);
    dstBuf[1] = (int*)carve(sizeof(int) * ETOT);
    emBuf[1]  = (int*)carve(sizeof(int) * ETOT);

    const int E = ETOT;
    dim3 eb((E + 255) / 256), tb(256);

    // initial edge state: raw edge_index rows, mask = all ones (in buf 0)
    fill_int<<<eb, tb, 0, stream>>>(emBuf[0], 1, E);
    const int* srcCur = edge_index;
    const int* dstCur = edge_index + E;
    const int* emCur  = emBuf[0];
    int nextIdx = 1;   // buffer set receiving the remapped edges

    const float* xin = x_in;
    int n = N_PER;

    for (int blk = 0; blk < 3; blk++) {
        const float* att_W     = (const float*)d_in[3 + 7 * blk + 0];
        const float* att_a_dst = (const float*)d_in[3 + 7 * blk + 1];
        const float* att_a_src = (const float*)d_in[3 + 7 * blk + 2];
        const float* att_b     = (const float*)d_in[3 + 7 * blk + 3];
        const float* lin_W     = (const float*)d_in[3 + 7 * blk + 4];
        const float* lin_b     = (const float*)d_in[3 + 7 * blk + 5];
        const float* pool_p    = (const float*)d_in[3 + 7 * blk + 6];

        int N = BGRAPH * n;
        int k = n / 2;  // ceil(0.5*n), n always even

        // h = x @ att_W.T  [N, 512]
        gemm_wmma<<<dim3(HC / TN, N / TM), dim3(128), 0, stream>>>(
            xin, att_W, nullptr, h, N, HC, HID, 0);

        alpha_kernel<<<dim3(N), dim3(128), 0, stream>>>(h, att_a_src, att_a_dst, as_n, ad_n);

        // CSR by destination
        fill_int<<<dim3((N + 255) / 256), tb, 0, stream>>>(cnt, 0, N);
        csr_count<<<eb, tb, 0, stream>>>(dstCur, emCur, E, cnt);
        csr_scan<<<dim3(BGRAPH), dim3(256), 0, stream>>>(cnt, offs, cursor, n);
        csr_fill<<<eb, tb, 0, stream>>>(dstCur, emCur, E, cursor, elist);

        gat_aggregate<<<dim3(N), dim3(128), 0, stream>>>(
            h, as_n, ad_n, att_b, srcCur, offs, cnt, elist, gatout);

        // linout = gatout @ lin_W.T + lin_b  [N, 128]
        gemm_wmma<<<dim3(HID / TN, N / TM), dim3(128), 0, stream>>>(
            gatout, lin_W, lin_b, linout, N, HID, HC, 0);

        // TopK pooling
        pnorm_kernel<<<dim3(1), dim3(128), 0, stream>>>(pool_p, pnorm);
        score_kernel<<<dim3((N + 3) / 4), dim3(128), 0, stream>>>(linout, pool_p, pnorm, score, N);
        topk_kernel<<<dim3(BGRAPH), dim3(256), 0, stream>>>(score, n, k, perm, new_pos);

        float* xout = (blk & 1) ? xB : xA;
        int NK = BGRAPH * k;
        gather_scale<<<dim3((NK * HID + 255) / 256), tb, 0, stream>>>(
            linout, score, perm, xout, NK);

        edge_remap<<<eb, tb, 0, stream>>>(srcCur, dstCur, emCur, new_pos, E,
                                          srcBuf[nextIdx], dstBuf[nextIdx], emBuf[nextIdx]);

        readout_kernel<<<dim3(BGRAPH), dim3(128), 0, stream>>>(xout, k, z, blk == 0);

        // rotate edge buffers
        srcCur = srcBuf[nextIdx];
        dstCur = dstBuf[nextIdx];
        emCur  = emBuf[nextIdx];
        nextIdx ^= 1;

        xin = xout;
        n = k;
    }

    // FC head: z[128,256] -> relu fc1 -> relu fc2 -> fc3
    gemm_wmma<<<dim3(FC1 / TN, BGRAPH / TM), dim3(128), 0, stream>>>(
        z, fc_W1, fc_b1, fc1o, BGRAPH, FC1, 2 * HID, 1);
    gemm_wmma<<<dim3(FC2 / TN, BGRAPH / TM), dim3(128), 0, stream>>>(
        fc1o, fc_W2, fc_b2, fc2o, BGRAPH, FC2, FC1, 1);
    fc3_kernel<<<dim3(BGRAPH), dim3(32), 0, stream>>>(fc2o, fc_W3, fc_b3, (float*)d_out);
}
